// MamMLP_25847113187506
// MI455X (gfx1250) — compile-verified, matrix operand order checked by reading, not run on hardware
//
#include <hip/hip_runtime.h>
#include <hip/hip_bf16.h>
#include <math.h>

// ---------------------------------------------------------------------------
// MamMLP forward for MI455X (gfx1250, wave32, WMMA).
// GEMMs: v_wmma_f32_16x16x32_bf16 (bf16 in, fp32 accumulate), LDS staged via
// CDNA5 async Global->LDS copies (ASYNCcnt) with double buffering.
// Selective scan: register-resident recurrence over L with 16 states.
// ---------------------------------------------------------------------------

typedef __attribute__((ext_vector_type(16))) __bf16 v16bf;
typedef __attribute__((ext_vector_type(8)))  float  v8f;

constexpr int Bn  = 2;
constexpr int Ln  = 1024;
constexpr int DMn = 1024;
constexpr int DSn = 16;
constexpr int DFFn= 4096;
constexpr int DIn = 2048;
constexpr int DTRn= 64;
constexpr int KCn = 4;
constexpr int NTn = Bn * Ln;          // 2048 tokens
constexpr int XDBL= DTRn + 2 * DSn;   // 96

// --- CDNA5 async Global->LDS copy (per-lane, 16B), tracked by ASYNCcnt -----
__device__ __forceinline__ void async_copy_b128(unsigned lds_off, const void* gaddr) {
  asm volatile("global_load_async_to_lds_b128 %0, %1, off"
               :
               : "v"(lds_off), "v"(gaddr)
               : "memory");
}
__device__ __forceinline__ void wait_async0() {
#if defined(__has_builtin)
#if __has_builtin(__builtin_amdgcn_s_wait_asynccnt)
  __builtin_amdgcn_s_wait_asynccnt(0);
#else
  asm volatile("s_wait_asynccnt 0x0" ::: "memory");
#endif
#else
  asm volatile("s_wait_asynccnt 0x0" ::: "memory");
#endif
}

// ---------------------------------------------------------------------------
// fp32 -> bf16 conversion (grid-stride)
// ---------------------------------------------------------------------------
__global__ __launch_bounds__(256) void cvt_f32_bf16(const float* __restrict__ in,
                                                    __bf16* __restrict__ out, int n) {
  for (int i = blockIdx.x * blockDim.x + threadIdx.x; i < n;
       i += gridDim.x * blockDim.x)
    out[i] = (__bf16)in[i];
}

// ---------------------------------------------------------------------------
// Tiled WMMA GEMM:  C(M,N) = A(M,K) * W(N,K)^T  [+ bias / activation]
//   Workgroup: 256 threads = 8 waves.  WG tile 128(M) x 64(N), K-step 32.
//   Waves arranged 4(M) x 2(N); each wave computes 32x32 via 2x2 WMMA frags.
//   LDS row stride 56 bf16 (112B): 16B-aligned b128 LDS ops, conflict-free
//   fragment reads (28*m mod 64 distinct for m=0..15).
//   Double-buffered LDS, filled by async Global->LDS (one barrier / K-step).
// epi: 0=none, 1=softplus(v+bias), 2=gelu(v+bias) exact, 3=v+bias
// ---------------------------------------------------------------------------
union FragAB { v16bf v; uint4 u[2]; };

__global__ __launch_bounds__(256) void wmma_gemm_bf16(
    const __bf16* __restrict__ A, int lda,
    const __bf16* __restrict__ W, int ldw,
    const float* __restrict__ bias,
    float* __restrict__ Cf, __bf16* __restrict__ Cb,
    int M, int N, int K, int epi)
{
  constexpr int TKP = 56;                 // padded K stride in elements
  __shared__ __bf16 As[2][128][TKP];
  __shared__ __bf16 Bs[2][64][TKP];

  const int tid  = threadIdx.x;
  const int lane = tid & 31;
  const int wid  = tid >> 5;
  const int wm   = wid & 3;               // wave row   (4 along M)
  const int wn   = wid >> 2;              // wave col   (2 along N)
  const int hl   = lane >> 4;             // lane half
  const int ml   = lane & 15;
  const int m0   = blockIdx.x * 128;
  const int n0   = blockIdx.y * 64;

  // per-thread staging coordinates (one 16B chunk each)
  const int arow0 = tid >> 2, aquad0 = tid & 3;           // A chunks 0..255
  const int arow1 = (tid + 256) >> 2, aquad1 = tid & 3;   // A chunks 256..511
  const int brow  = tid >> 2, bquad = tid & 3;            // B chunks 0..255
  int wr = n0 + brow; if (wr > N - 1) wr = N - 1;         // clamp for N=96

  auto stage = [&](int buf, int kk) {
    async_copy_b128((unsigned)(size_t)&As[buf][arow0][aquad0 * 8],
                    A + (size_t)(m0 + arow0) * lda + kk + aquad0 * 8);
    async_copy_b128((unsigned)(size_t)&As[buf][arow1][aquad1 * 8],
                    A + (size_t)(m0 + arow1) * lda + kk + aquad1 * 8);
    async_copy_b128((unsigned)(size_t)&Bs[buf][brow][bquad * 8],
                    W + (size_t)wr * ldw + kk + bquad * 8);
  };

  v8f acc[2][2];
  #pragma unroll
  for (int i = 0; i < 2; ++i)
    #pragma unroll
    for (int j = 0; j < 2; ++j)
      acc[i][j] = (v8f){0.f,0.f,0.f,0.f,0.f,0.f,0.f,0.f};

  stage(0, 0);                            // prologue fill of buffer 0

  const int nsteps = K >> 5;
  for (int s = 0; s < nsteps; ++s) {
    const int buf = s & 1;
    wait_async0();                        // this wave's async copies done
    __syncthreads();                      // tile visible to all 8 waves;
                                          // also retires readers of buf^1
    if (s + 1 < nsteps) {
      stage(buf ^ 1, (s + 1) << 5);       // overlap next copy with compute
      __builtin_prefetch(A + (size_t)(m0 + arow0) * lda + ((s + 2) << 5), 0, 0);
      __builtin_prefetch(W + (size_t)wr * ldw + ((s + 2) << 5), 0, 0);
    }

    // ---- build fragments per documented 16-bit VGPR layout ----
    // A 16x32: lane m holds row m; vector elems [0..7] <- k in [h*8, h*8+8),
    //          elems [8..15] <- k in [16+h*8, 16+h*8+8).
    FragAB af[2], bf2[2];
    #pragma unroll
    for (int i = 0; i < 2; ++i) {
      int ra = wm * 32 + i * 16 + ml;
      af[i].u[0]  = *(const uint4*)&As[buf][ra][hl * 8];
      af[i].u[1]  = *(const uint4*)&As[buf][ra][16 + hl * 8];
      int rb = wn * 32 + i * 16 + ml;
      bf2[i].u[0] = *(const uint4*)&Bs[buf][rb][hl * 8];
      bf2[i].u[1] = *(const uint4*)&Bs[buf][rb][16 + hl * 8];
    }
    #pragma unroll
    for (int i = 0; i < 2; ++i)
      #pragma unroll
      for (int j = 0; j < 2; ++j)
        acc[i][j] = __builtin_amdgcn_wmma_f32_16x16x32_bf16(
            false, af[i].v, false, bf2[j].v, (short)0, acc[i][j], false, false);
  }

  // ---- epilogue + store (C/D layout: lane h*16+n holds M = r + 8*h) ----
  #pragma unroll
  for (int i = 0; i < 2; ++i) {
    #pragma unroll
    for (int j = 0; j < 2; ++j) {
      int gcol = n0 + wn * 32 + j * 16 + ml;
      if (gcol >= N) continue;
      float bv = (epi != 0 && bias != nullptr) ? bias[gcol] : 0.f;
      #pragma unroll
      for (int r = 0; r < 8; ++r) {
        int grow = m0 + wm * 32 + i * 16 + hl * 8 + r;
        float v  = acc[i][j][r];
        if (epi == 1)      { v += bv; v = (v > 20.f) ? v : log1pf(expf(v)); }
        else if (epi == 2) { v += bv; v = 0.5f * v * (1.f + erff(v * 0.70710678118f)); }
        else if (epi == 3) { v += bv; }
        size_t o = (size_t)grow * N + gcol;
        if (Cf) Cf[o] = v;
        if (Cb) Cb[o] = (__bf16)v;
      }
    }
  }
}

// ---------------------------------------------------------------------------
// Causal depthwise conv (K=4) + SiLU over x = xz[:, :, 0:DI]
// one thread per (b,l,d); writes fp32 (for the scan) and bf16 (for x_proj).
// ---------------------------------------------------------------------------
__global__ __launch_bounds__(256) void conv_silu(
    const float* __restrict__ xz, const float* __restrict__ cw,
    const float* __restrict__ cb, float* __restrict__ xo,
    __bf16* __restrict__ xob)
{
  int g = blockIdx.x * blockDim.x + threadIdx.x;      // 0 .. NT*DI-1
  int d = g % DIn;
  int t = g / DIn;                                    // token = b*L + l
  int l = t % Ln;
  float acc = cb[d];
  #pragma unroll
  for (int j = 0; j < KCn; ++j) {
    int ls = l - (KCn - 1) + j;
    if (ls >= 0)
      acc += xz[(size_t)(t - (KCn - 1) + j) * (2 * DIn) + d] * cw[d * KCn + j];
  }
  float s = acc / (1.f + expf(-acc));                 // silu
  xo[(size_t)t * DIn + d]  = s;
  xob[(size_t)t * DIn + d] = (__bf16)s;
}

// ---------------------------------------------------------------------------
// Selective scan: one thread per (b,d); 16 states live in registers.
//   h[n] = exp(dt*A[d][n])*h[n] + dt*B[l][n]*x ;  y = sum_n h[n]*C[l][n]
//   out  = (y + x*D[d]) * silu(z)   -> bf16 (feeds out_proj GEMM)
// ---------------------------------------------------------------------------
__global__ __launch_bounds__(256) void mamba_scan(
    const float* __restrict__ dtp, const float* __restrict__ xc,
    const float* __restrict__ xdbl, const float* __restrict__ xz,
    const float* __restrict__ A_log, const float* __restrict__ Dp,
    __bf16* __restrict__ ybf)
{
  int g = blockIdx.x * blockDim.x + threadIdx.x;      // 0 .. Bn*DIn-1
  int d = g % DIn;
  int b = g / DIn;

  float An[DSn], h[DSn];
  #pragma unroll
  for (int n = 0; n < DSn; ++n) {
    An[n] = -expf(A_log[(size_t)d * DSn + n]);        // A = -exp(A_log)
    h[n]  = 0.f;
  }
  const float Dd = Dp[d];

  for (int l = 0; l < Ln; ++l) {
    const int t = b * Ln + l;
    const float dtv = dtp[(size_t)t * DIn + d];
    const float xv  = xc [(size_t)t * DIn + d];
    const float zv  = xz [(size_t)t * (2 * DIn) + DIn + d];
    const float* Bm = xdbl + (size_t)t * XDBL + DTRn;        // 16 values
    const float* Cm = xdbl + (size_t)t * XDBL + DTRn + DSn;  // 16 values
    float acc = 0.f;
    #pragma unroll
    for (int n = 0; n < DSn; ++n) {
      float dA = __expf(dtv * An[n]);
      h[n] = dA * h[n] + dtv * Bm[n] * xv;
      acc += h[n] * Cm[n];
    }
    float y = acc + xv * Dd;
    float gz = y * (zv / (1.f + expf(-zv)));                 // * silu(z)
    ybf[(size_t)t * DIn + d] = (__bf16)gz;
  }
}

// ---------------------------------------------------------------------------
// out = rmsnorm(a + b) * w  ; one block (256 thr) per token row of 1024
// ---------------------------------------------------------------------------
__global__ __launch_bounds__(256) void add_rmsnorm(
    const float* __restrict__ a, const float* __restrict__ b,
    const float* __restrict__ w, float* __restrict__ of,
    __bf16* __restrict__ ob)
{
  __shared__ float red[256];
  const int t   = blockIdx.x;
  const int tid = threadIdx.x;
  float v[4];
  float ss = 0.f;
  #pragma unroll
  for (int i = 0; i < 4; ++i) {
    int c = tid + i * 256;
    v[i] = a[(size_t)t * DMn + c] + b[(size_t)t * DMn + c];
    ss += v[i] * v[i];
  }
  red[tid] = ss;
  __syncthreads();
  for (int s = 128; s > 0; s >>= 1) {
    if (tid < s) red[tid] += red[tid + s];
    __syncthreads();
  }
  const float rs = rsqrtf(red[0] / (float)DMn + 1e-6f);
  #pragma unroll
  for (int i = 0; i < 4; ++i) {
    int c = tid + i * 256;
    float o = v[i] * rs * w[c];
    of[(size_t)t * DMn + c] = o;
    if (ob) ob[(size_t)t * DMn + c] = (__bf16)o;
  }
}

// ---------------------------------------------------------------------------
// host launcher
// ---------------------------------------------------------------------------
extern "C" void kernel_launch(void* const* d_in, const int* in_sizes, int n_in,
                              void* d_out, int out_size, void* d_ws, size_t ws_size,
                              hipStream_t stream) {
  (void)in_sizes; (void)n_in; (void)out_size; (void)ws_size;

  const float* Z      = (const float*)d_in[0];
  const float* in_w   = (const float*)d_in[1];
  const float* conv_w = (const float*)d_in[2];
  const float* conv_b = (const float*)d_in[3];
  const float* xp_w   = (const float*)d_in[4];
  const float* dt_w   = (const float*)d_in[5];
  const float* dt_b   = (const float*)d_in[6];
  const float* A_log  = (const float*)d_in[7];
  const float* Dp     = (const float*)d_in[8];
  const float* out_w  = (const float*)d_in[9];
  const float* w1     = (const float*)d_in[10];
  const float* b1     = (const float*)d_in[11];
  const float* w2     = (const float*)d_in[12];
  const float* b2     = (const float*)d_in[13];
  const float* nw     = (const float*)d_in[14];
  float* outp = (float*)d_out;

  char* ws = (char*)d_ws;
  size_t off = 0;
  auto carve = [&](size_t bytes) -> void* {
    void* p = ws + off;
    off += (bytes + 255) & ~(size_t)255;
    return p;
  };

  __bf16* Zbf    = (__bf16*)carve((size_t)NTn * DMn * 2);
  __bf16* w_inb  = (__bf16*)carve((size_t)2 * DIn * DMn * 2);
  __bf16* w_xpb  = (__bf16*)carve((size_t)XDBL * DIn * 2);
  __bf16* w_dtb  = (__bf16*)carve((size_t)DIn * DTRn * 2);
  __bf16* w_outb = (__bf16*)carve((size_t)DMn * DIn * 2);
  __bf16* w_1b   = (__bf16*)carve((size_t)DFFn * DMn * 2);
  __bf16* w_2b   = (__bf16*)carve((size_t)DMn * DFFn * 2);
  float*  xz     = (float*) carve((size_t)NTn * 2 * DIn * 4);
  float*  xcv    = (float*) carve((size_t)NTn * DIn * 4);
  __bf16* xbf    = (__bf16*)carve((size_t)NTn * DIn * 2);
  float*  xdbl   = (float*) carve((size_t)NTn * XDBL * 4);
  __bf16* xdblb  = (__bf16*)carve((size_t)NTn * XDBL * 2);
  float*  dtf    = (float*) carve((size_t)NTn * DIn * 4);
  __bf16* ybf    = (__bf16*)carve((size_t)NTn * DIn * 2);
  float*  zmam   = (float*) carve((size_t)NTn * DMn * 4);
  __bf16* zmamb  = (__bf16*)carve((size_t)NTn * DMn * 2);
  // buffer reuse (stream-ordered, so safe):
  float*  outm = xcv;          // out_proj result reuses conv-x (dead after scan)
  __bf16* hbf  = (__bf16*)dtf; // mlp hidden reuses dt (dead after scan)
  float*  mlpo = xz;           // mlp output reuses xz (z dead after scan)

  auto cvt = [&](const float* s, __bf16* dst, int n) {
    int blocks = (n + 255) / 256; if (blocks > 4096) blocks = 4096;
    cvt_f32_bf16<<<blocks, 256, 0, stream>>>(s, dst, n);
  };
  auto gemm = [&](const __bf16* A, int lda, const __bf16* Wm, int ldw,
                  const float* bias, float* Cf, __bf16* Cb,
                  int M, int N, int K, int epi) {
    dim3 g(M / 128, (N + 63) / 64);
    wmma_gemm_bf16<<<g, dim3(256), 0, stream>>>(A, lda, Wm, ldw, bias, Cf, Cb,
                                                M, N, K, epi);
  };

  // ---- weight / input conversions to bf16 ----
  cvt(Z,     Zbf,    NTn * DMn);
  cvt(in_w,  w_inb,  2 * DIn * DMn);
  cvt(xp_w,  w_xpb,  XDBL * DIn);
  cvt(dt_w,  w_dtb,  DIn * DTRn);
  cvt(out_w, w_outb, DMn * DIn);
  cvt(w1,    w_1b,   DFFn * DMn);
  cvt(w2,    w_2b,   DMn * DFFn);

  // ---- 1. in_proj: xz = Z @ in_proj_w^T  (2048 x 4096) ----
  gemm(Zbf, DMn, w_inb, DMn, nullptr, xz, nullptr, NTn, 2 * DIn, DMn, 0);

  // ---- 2. causal depthwise conv + silu ----
  conv_silu<<<(NTn * DIn) / 256, 256, 0, stream>>>(xz, conv_w, conv_b, xcv, xbf);

  // ---- 3. x_proj: x_dbl = x @ x_proj_w^T  (2048 x 96) ----
  gemm(xbf, DIn, w_xpb, DIn, nullptr, xdbl, nullptr, NTn, XDBL, DIn, 0);
  cvt(xdbl, xdblb, NTn * XDBL);

  // ---- 4. dt = softplus(dt_part @ dt_proj_w^T + dt_b)  (2048 x 2048) ----
  gemm(xdblb, XDBL, w_dtb, DTRn, dt_b, dtf, nullptr, NTn, DIn, DTRn, 1);

  // ---- 5. selective scan + D skip + silu(z) gate -> y (bf16) ----
  mamba_scan<<<(Bn * DIn) / 256, 256, 0, stream>>>(dtf, xcv, xdbl, xz,
                                                   A_log, Dp, ybf);

  // ---- 6. out_proj: y @ out_proj_w^T  (2048 x 1024) ----
  gemm(ybf, DIn, w_outb, DIn, nullptr, outm, nullptr, NTn, DMn, DIn, 0);

  // ---- 7. Z_mam = rmsnorm(out_mamba + Z) * norm1_w ----
  add_rmsnorm<<<NTn, 256, 0, stream>>>(outm, Z, nw, zmam, zmamb);

  // ---- 8. MLP up: h = gelu(Z_mam @ w1^T + b1)  (2048 x 4096, bf16) ----
  gemm(zmamb, DMn, w_1b, DMn, b1, nullptr, hbf, NTn, DFFn, DMn, 2);

  // ---- 9. MLP down: mlp = h @ w2^T + b2  (2048 x 1024) ----
  gemm(hbf, DFFn, w_2b, DFFn, b2, mlpo, nullptr, NTn, DMn, DFFn, 3);

  // ---- 10. out = rmsnorm(mlp + Z_mam) * norm1_w ----
  add_rmsnorm<<<NTn, 256, 0, stream>>>(mlpo, zmam, nw, outp, nullptr);
}
